// selfAttentionV_26869315404644
// MI455X (gfx1250) — matrix operand order repaired and checked by reading, not compiled, not run on hardware
//
#include <hip/hip_runtime.h>
#include <hip/hip_bf16.h>

// ---------------- problem constants ----------------
#define BATCH   512
#define SEQ     50
#define HID     2048
#define MROWS   (BATCH * SEQ)          // 25600
#define NPTS    (BATCH * SEQ * SEQ)    // 1,280,000 points (2-D)
#define WPLEN   (BATCH * SEQ * 2 * SEQ)// 2,560,000 floats
#define KM_IT   10
#define BK      32
#define LDSW    40                     // padded LDS row (80B): bank-conflict-free b128 reads

typedef __attribute__((ext_vector_type(16))) __bf16 v16bf;
typedef __attribute__((ext_vector_type(8)))  __bf16 bf16x8;
typedef __attribute__((ext_vector_type(4)))  __bf16 bf16x4;
typedef __attribute__((ext_vector_type(8)))  float  v8f;

static __device__ __forceinline__ v16bf join16(bf16x8 lo, bf16x8 hi) {
    return __builtin_shufflevector(lo, hi, 0, 1, 2, 3, 4, 5, 6, 7,
                                           8, 9, 10, 11, 12, 13, 14, 15);
}

// ---------------- fp32 -> bf16 convert (no transpose needed: W is [N][K] row-major) ------
__global__ __launch_bounds__(256) void convert_w_kernel(const float* __restrict__ in,
                                                        __bf16* __restrict__ out) {
    size_t i = ((size_t)blockIdx.x * 256 + threadIdx.x) * 8;
    float4 f0 = *reinterpret_cast<const float4*>(in + i);
    float4 f1 = *reinterpret_cast<const float4*>(in + i + 4);
    bf16x8 v = { (__bf16)f0.x, (__bf16)f0.y, (__bf16)f0.z, (__bf16)f0.w,
                 (__bf16)f1.x, (__bf16)f1.y, (__bf16)f1.z, (__bf16)f1.w };
    *reinterpret_cast<bf16x8*>(out + i) = v;
}

// ---------------- projection: Out = A @ W^T + bias + pe[b]  (bf16 out) ----------------
// A: [MROWS][HID] fp32, Wbf: [HID(N)][HID(K)] bf16 (row-major, as given), Out: bf16
__global__ __launch_bounds__(128) void proj_kernel(const float* __restrict__ A,
                                                   const __bf16* __restrict__ Wbf,
                                                   const float* __restrict__ bias,
                                                   const float* __restrict__ pe,
                                                   __bf16* __restrict__ Out) {
    __shared__ __bf16 As[64][LDSW];   // M-major: As[m][k]
    __shared__ __bf16 Bs[64][LDSW];   // N-major: Bs[n][k]

    const int tid  = threadIdx.x;
    const int lane = tid & 31;
    const int wave = tid >> 5;              // 0..3, owns M-strip wave*16
    const int mBase = blockIdx.x * 64;
    const int nBase = blockIdx.y * 64;

    v8f acc[4] = {};                        // 4 tiles of 16x16 along N

    for (int k0 = 0; k0 < HID; k0 += BK) {
        // prefetch next K tile (global_prefetch_b8 path)
        if (k0 + BK < HID) {
            if (tid < 64) __builtin_prefetch(&A  [(size_t)(mBase + tid)        * HID + k0 + BK], 0, 1);
            else          __builtin_prefetch(&Wbf[(size_t)(nBase + (tid - 64)) * HID + k0 + BK], 0, 1);
        }
        // stage A tile 64x32 (fp32 -> bf16), vectorized float4 -> bf16x4
        #pragma unroll
        for (int i = 0; i < 4; ++i) {               // 512 float4-chunks / 128 threads
            int idx = i * 128 + tid;
            int r = idx >> 3, c = (idx & 7) * 4;
            float4 f = *reinterpret_cast<const float4*>(
                &A[(size_t)(mBase + r) * HID + (k0 + c)]);
            bf16x4 v = { (__bf16)f.x, (__bf16)f.y, (__bf16)f.z, (__bf16)f.w };
            *reinterpret_cast<bf16x4*>(&As[r][c]) = v;
        }
        // stage B tile 64x32 (bf16 16B-chunk copies)
        #pragma unroll
        for (int i = 0; i < 2; ++i) {               // 256 16B-chunks / 128 threads
            int idx = i * 128 + tid;
            int r = idx >> 2, c = (idx & 3) * 8;
            *reinterpret_cast<uint4*>(&Bs[r][c]) = *reinterpret_cast<const uint4*>(
                &Wbf[(size_t)(nBase + r) * HID + (k0 + c)]);
        }
        __syncthreads();

        // A fragment: lane&15 = M row; lane half selects K 0-15 vs 16-31 (contiguous)
        const int arow = wave * 16 + (lane & 15);
        const int koff = (lane >> 4) * 16;
        v16bf afrag = join16(*reinterpret_cast<const bf16x8*>(&As[arow][koff]),
                             *reinterpret_cast<const bf16x8*>(&As[arow][koff + 8]));

        // B fragment: lane&15 = N col; lane half selects K 0-15 vs 16-31 (contiguous)
        const int bcol  = lane & 15;
        const int koff2 = (lane >> 4) * 16;
        #pragma unroll
        for (int t = 0; t < 4; ++t) {
            const __bf16* bp = &Bs[t * 16 + bcol][koff2];
            v16bf bfrag = join16(*reinterpret_cast<const bf16x8*>(bp),
                                 *reinterpret_cast<const bf16x8*>(bp + 8));
            acc[t] = __builtin_amdgcn_wmma_f32_16x16x32_bf16(
                false, afrag, false, bfrag, (short)0, acc[t], false, false);
        }
        __syncthreads();
    }

    // epilogue: C/D layout -> + bias + pe[b], store bf16
    const int mLocal = wave * 16 + ((lane >> 4) << 3);  // +8 for upper lane half
    #pragma unroll
    for (int t = 0; t < 4; ++t) {
        #pragma unroll
        for (int r = 0; r < 8; ++r) {
            int m = mBase + mLocal + r;
            int n = nBase + t * 16 + (lane & 15);
            int b = m / SEQ;
            float v = acc[t][r] + bias[n] + pe[(size_t)b * HID + n];
            Out[(size_t)m * HID + n] = (__bf16)v;
        }
    }
}

// ---------------- per-batch: score = Q K^T, softmax, wp = prob @ lin_w^T + lin_b ----------
__global__ __launch_bounds__(256) void score_kernel(const __bf16* __restrict__ Q,
                                                    const __bf16* __restrict__ K,
                                                    const float* __restrict__ lin_w,
                                                    const float* __restrict__ lin_b,
                                                    float* __restrict__ wp) {
    __shared__ __bf16 Qs[64][LDSW];   // row q, K contiguous
    __shared__ __bf16 Ks[64][LDSW];   // row n, K contiguous (B operand, N-major)
    __shared__ float  Ss[64][64];     // 16 KB

    const int b    = blockIdx.x;
    const int tid  = threadIdx.x;
    const int lane = tid & 31;
    const int wave = tid >> 5;              // 8 waves -> 2 tiles each (4x4 tile grid)
    const int tr   = wave & 3;              // tile row
    const int tc0  = (wave >> 2) * 2;       // first tile col

    const __bf16* Qb = Q + (size_t)b * SEQ * HID;
    const __bf16* Kb = K + (size_t)b * SEQ * HID;

    v8f acc[2] = {};

    for (int k0 = 0; k0 < HID; k0 += BK) {
        if (k0 + BK < HID && tid < SEQ * 2) {
            const __bf16* base = (tid < SEQ) ? Qb : Kb;
            int row = (tid < SEQ) ? tid : (tid - SEQ);
            __builtin_prefetch(&base[(size_t)row * HID + k0 + BK], 0, 1);
        }
        // stage Q and K tiles: identical coalesced 16B copies, zero rows >= 50
        #pragma unroll
        for (int i = 0; i < 2; ++i) {               // 256 16B-chunks each, 256 threads
            int idx = i * 256 + tid;
            int r = (idx & 255) >> 2, c = (idx & 3) * 8;
            uint4 zv = make_uint4(0u, 0u, 0u, 0u);
            if (i == 0) {
                *reinterpret_cast<uint4*>(&Qs[r][c]) = (r < SEQ)
                    ? *reinterpret_cast<const uint4*>(&Qb[(size_t)r * HID + (k0 + c)]) : zv;
            } else {
                *reinterpret_cast<uint4*>(&Ks[r][c]) = (r < SEQ)
                    ? *reinterpret_cast<const uint4*>(&Kb[(size_t)r * HID + (k0 + c)]) : zv;
            }
        }
        __syncthreads();

        const int arow = tr * 16 + (lane & 15);
        const int koff = (lane >> 4) * 16;
        v16bf afrag = join16(*reinterpret_cast<const bf16x8*>(&Qs[arow][koff]),
                             *reinterpret_cast<const bf16x8*>(&Qs[arow][koff + 8]));

        const int bcol  = lane & 15;
        const int koff2 = (lane >> 4) * 16;
        #pragma unroll
        for (int t = 0; t < 2; ++t) {
            const __bf16* bp = &Ks[(tc0 + t) * 16 + bcol][koff2];
            v16bf bfrag = join16(*reinterpret_cast<const bf16x8*>(bp),
                                 *reinterpret_cast<const bf16x8*>(bp + 8));
            acc[t] = __builtin_amdgcn_wmma_f32_16x16x32_bf16(
                false, afrag, false, bfrag, (short)0, acc[t], false, false);
        }
        __syncthreads();
    }

    // spill scores to LDS
    const int mLocal = tr * 16 + ((lane >> 4) << 3);
    #pragma unroll
    for (int t = 0; t < 2; ++t) {
        #pragma unroll
        for (int r = 0; r < 8; ++r)
            Ss[mLocal + r][(tc0 + t) * 16 + (lane & 15)] = acc[t][r];
    }
    __syncthreads();

    // softmax over valid 50x50 region (one row per thread)
    if (tid < SEQ) {
        float mx = -3.402823466e+38f;
        for (int c = 0; c < SEQ; ++c) mx = fmaxf(mx, Ss[tid][c]);
        float sum = 0.f;
        for (int c = 0; c < SEQ; ++c) { float e = __expf(Ss[tid][c] - mx); Ss[tid][c] = e; sum += e; }
        float inv = 1.f / sum;
        for (int c = 0; c < SEQ; ++c) Ss[tid][c] *= inv;
    }
    __syncthreads();

    // wp[b, q, j] = sum_s prob[q,s] * lin_w[j,s] + lin_b[j]   (50 x 100 outputs)
    for (int o = tid; o < SEQ * 2 * SEQ; o += 256) {
        int q = o / (2 * SEQ);
        int j = o - q * (2 * SEQ);
        float s = lin_b[j];
        for (int c = 0; c < SEQ; ++c) s += Ss[q][c] * lin_w[j * SEQ + c];
        wp[(size_t)b * (SEQ * 2 * SEQ) + o] = s;
    }
}

// ---------------- k-means (k=2, 2-D), deterministic two-stage reduction ----------------
__global__ void km_init_kernel(const float* __restrict__ wp, float* __restrict__ cent) {
    if (blockIdx.x == 0 && threadIdx.x == 0) {
        cent[0] = wp[0]; cent[1] = wp[1];   // centroid 0 = points[0]
        cent[2] = wp[2]; cent[3] = wp[3];   // centroid 1 = points[1]
    }
}

__global__ __launch_bounds__(256) void km_reduce_kernel(const float* __restrict__ wp,
                                                        const float* __restrict__ cent,
                                                        float* __restrict__ partials) {
    __shared__ float sd[6][256];
    const int tid = threadIdx.x;
    const int p   = blockIdx.x * 256 + tid;           // NPTS == gridDim*256 exactly
    const float x = wp[2 * p], y = wp[2 * p + 1];
    const float dx0 = x - cent[0], dy0 = y - cent[1];
    const float dx1 = x - cent[2], dy1 = y - cent[3];
    const float d0 = dx0 * dx0 + dy0 * dy0;
    const float d1 = dx1 * dx1 + dy1 * dy1;
    const int   l  = (d0 <= d1) ? 0 : 1;              // argmin, first-index ties
    sd[0][tid] = (l == 0) ? x   : 0.f;
    sd[1][tid] = (l == 0) ? y   : 0.f;
    sd[2][tid] = (l == 0) ? 1.f : 0.f;
    sd[3][tid] = (l == 1) ? x   : 0.f;
    sd[4][tid] = (l == 1) ? y   : 0.f;
    sd[5][tid] = (l == 1) ? 1.f : 0.f;
    __syncthreads();
    for (int s = 128; s > 0; s >>= 1) {
        if (tid < s) {
            #pragma unroll
            for (int i = 0; i < 6; ++i) sd[i][tid] += sd[i][tid + s];
        }
        __syncthreads();
    }
    if (tid == 0) {
        #pragma unroll
        for (int i = 0; i < 6; ++i) partials[blockIdx.x * 6 + i] = sd[i][0];
    }
}

__global__ __launch_bounds__(256) void km_update_kernel(const float* __restrict__ partials,
                                                        float* __restrict__ cent,
                                                        int nblocks) {
    __shared__ float sd[6][256];
    const int tid = threadIdx.x;
    float a[6] = {0.f, 0.f, 0.f, 0.f, 0.f, 0.f};
    for (int i = tid; i < nblocks; i += 256) {
        #pragma unroll
        for (int j = 0; j < 6; ++j) a[j] += partials[i * 6 + j];
    }
    #pragma unroll
    for (int j = 0; j < 6; ++j) sd[j][tid] = a[j];
    __syncthreads();
    for (int s = 128; s > 0; s >>= 1) {
        if (tid < s) {
            #pragma unroll
            for (int i = 0; i < 6; ++i) sd[i][tid] += sd[i][tid + s];
        }
        __syncthreads();
    }
    if (tid == 0) {
        float c0 = sd[2][0], c1 = sd[5][0];
        if (c0 > 0.f) { cent[0] = sd[0][0] / fmaxf(c0, 1.f); cent[1] = sd[1][0] / fmaxf(c0, 1.f); }
        if (c1 > 0.f) { cent[2] = sd[3][0] / fmaxf(c1, 1.f); cent[3] = sd[4][0] / fmaxf(c1, 1.f); }
    }
}

__global__ __launch_bounds__(256) void km_label_kernel(const float* __restrict__ wp,
                                                       const float* __restrict__ cent,
                                                       float* __restrict__ out) {
    const int p = blockIdx.x * 256 + threadIdx.x;
    const float x = wp[2 * p], y = wp[2 * p + 1];
    const float dx0 = x - cent[0], dy0 = y - cent[1];
    const float dx1 = x - cent[2], dy1 = y - cent[3];
    const float d0 = dx0 * dx0 + dy0 * dy0;
    const float d1 = dx1 * dx1 + dy1 * dy1;
    out[p] = (d0 <= d1) ? 0.f : 1.f;
}

// ---------------- launch ----------------
extern "C" void kernel_launch(void* const* d_in, const int* in_sizes, int n_in,
                              void* d_out, int out_size, void* d_ws, size_t ws_size,
                              hipStream_t stream) {
    const float* text_emb = (const float*)d_in[0];   // [512,50,2048]
    const float* emb      = (const float*)d_in[1];   // [512,50,2048]
    const float* Wq_w     = (const float*)d_in[2];   // [2048,2048]
    const float* Wq_b     = (const float*)d_in[3];   // [2048]
    const float* Wk_w     = (const float*)d_in[4];   // [2048,2048]
    const float* Wk_b     = (const float*)d_in[5];   // [2048]
    const float* lin_w    = (const float*)d_in[6];   // [100,50]
    const float* lin_b    = (const float*)d_in[7];   // [100]
    const float* pe       = (const float*)d_in[8];   // [5000,1,2048]
    float* out = (float*)d_out;

    // workspace layout
    char* ws = (char*)d_ws;
    size_t off = 0;
    __bf16* wqb = (__bf16*)(ws + off); off += (size_t)HID * HID * 2;        // 8.39 MB
    __bf16* wkb = (__bf16*)(ws + off); off += (size_t)HID * HID * 2;        // 8.39 MB
    __bf16* qbf = (__bf16*)(ws + off); off += (size_t)MROWS * HID * 2;      // 104.9 MB
    __bf16* kbf = (__bf16*)(ws + off); off += (size_t)MROWS * HID * 2;      // 104.9 MB
    float*  wp  = (float*)(ws + off);  off += (size_t)WPLEN * 4;            // 10.2 MB
    float*  par = (float*)(ws + off);  off += (size_t)(NPTS / 256) * 6 * 4; // 120 KB
    float*  cen = (float*)(ws + off);  off += 256;

    // 1) weight fp32 -> bf16 (layout kept: [N][K] row-major, used directly as B operand)
    const int cvtBlocks = (HID * HID) / (256 * 8);   // 2048
    convert_w_kernel<<<cvtBlocks, 256, 0, stream>>>(Wq_w, wqb);
    convert_w_kernel<<<cvtBlocks, 256, 0, stream>>>(Wk_w, wkb);

    // 2) projections (bf16 WMMA, fused bias + pe)
    dim3 pg(MROWS / 64, HID / 64);
    proj_kernel<<<pg, 128, 0, stream>>>(text_emb, wqb, Wq_b, pe, qbf);
    proj_kernel<<<pg, 128, 0, stream>>>(emb,      wkb, Wk_b, pe, kbf);

    // 3) per-batch attention score + softmax + linear
    score_kernel<<<BATCH, 256, 0, stream>>>(qbf, kbf, lin_w, lin_b, wp);

    // 4) k-means, 10 fixed iterations, deterministic reductions
    const int nblk = NPTS / 256;   // 5000
    km_init_kernel<<<1, 1, 0, stream>>>(wp, cen);
    for (int it = 0; it < KM_IT; ++it) {
        km_reduce_kernel<<<nblk, 256, 0, stream>>>(wp, cen, par);
        km_update_kernel<<<1, 256, 0, stream>>>(par, cen, nblk);
    }
    km_label_kernel<<<nblk, 256, 0, stream>>>(wp, cen, out);

    (void)in_sizes; (void)n_in; (void)out_size; (void)ws_size;
}
